// EGNNLayerPerturb_30983894073591
// MI455X (gfx1250) — compile-verified, matrix operand order checked by reading, not run on hardware
//
#include <hip/hip_runtime.h>

typedef __attribute__((ext_vector_type(16))) __bf16 v16bf;
typedef __attribute__((ext_vector_type(8)))  __bf16 v8bf;
typedef __attribute__((ext_vector_type(8)))  float  v8f;

__device__ __forceinline__ float silu_f(float v)    { return v / (1.0f + __expf(-v)); }
__device__ __forceinline__ float sigmoid_f(float v) { return 1.0f / (1.0f + __expf(-v)); }

// Convert 8 consecutive fp32 (16B-aligned) to 8 packed bf16
__device__ __forceinline__ v8bf cvt8(const float* __restrict__ p) {
  const float4* p4 = (const float4*)p;
  float4 f0 = p4[0], f1 = p4[1];
  v8bf r;
  r[0] = (__bf16)f0.x; r[1] = (__bf16)f0.y; r[2] = (__bf16)f0.z; r[3] = (__bf16)f0.w;
  r[4] = (__bf16)f1.x; r[5] = (__bf16)f1.y; r[6] = (__bf16)f1.z; r[7] = (__bf16)f1.w;
  return r;
}

__device__ __forceinline__ v16bf cat16(v8bf lo, v8bf hi) {
  return __builtin_shufflevector(lo, hi, 0,1,2,3,4,5,6,7,8,9,10,11,12,13,14,15);
}

// A/B fragment from fp32 global row (row pointer + intra-row element offset)
__device__ __forceinline__ v16bf frag_from_f32(const float* __restrict__ row, int ko) {
  return cat16(cvt8(row + ko), cvt8(row + ko + 16));
}
// A/B fragment from bf16 row (LDS or global), 16B-aligned chunks
__device__ __forceinline__ v16bf frag_from_bf16(const __bf16* __restrict__ row, int ko) {
  return cat16(*(const v8bf*)(row + ko), *(const v8bf*)(row + ko + 16));
}

__device__ __forceinline__ v8f wmma_bf16(v16bf a, v16bf b, v8f c) {
  return __builtin_amdgcn_wmma_f32_16x16x32_bf16(false, a, false, b, (short)0, c, false, false);
}

// ---------------------------------------------------------------------------
// Prep: transpose fp32 [K][Ncols] -> bf16 [Ncols][ldk], zero-padded K
// ---------------------------------------------------------------------------
__global__ void transpose_to_bf16(const float* __restrict__ src, __bf16* __restrict__ dst,
                                  int K, int Ncols, int ldk) {
  int idx = blockIdx.x * blockDim.x + threadIdx.x;
  int total = Ncols * ldk;
  if (idx >= total) return;
  int n = idx / ldk, k = idx % ldk;
  float v = (k < K) ? src[k * Ncols + n] : 0.0f;
  dst[idx] = (__bf16)v;
}

__global__ void init_buffers(float* __restrict__ m_aggr, long nm,
                             float* __restrict__ x_out, const float* __restrict__ x, int nx) {
  long i = (long)blockIdx.x * blockDim.x + threadIdx.x;
  if (i < nm) m_aggr[i] = 0.0f;
  if (i < nx) x_out[i] = x[i];
}

// ---------------------------------------------------------------------------
// Edge kernel: one wave32 per 16-edge tile
// ---------------------------------------------------------------------------
__global__ void __launch_bounds__(32)
egnn_edge_kernel(const float* __restrict__ h, const float* __restrict__ x,
                 const int* __restrict__ ei, const float* __restrict__ emask,
                 const __bf16* __restrict__ We1T, const float* __restrict__ be1,
                 const __bf16* __restrict__ We2T, const float* __restrict__ be2,
                 const float* __restrict__ Wg,  const float* __restrict__ bg,
                 const __bf16* __restrict__ Wc1T, const float* __restrict__ bc1,
                 const float* __restrict__ Wc2, const float* __restrict__ bc2,
                 float* __restrict__ m_aggr, float* __restrict__ x_out, int E)
{
  __shared__ __bf16 sA[16 * 64];
  __shared__ int    sCol[16];
  __shared__ float  sCw[16];

  const int l  = threadIdx.x;
  const int hi = l >> 4;            // 0: lanes 0-15, 1: lanes 16-31
  const int lr = l & 15;
  const int ko0 = hi ? 8 : 0;       // base K offset inside a 32-K block
  const int base = blockIdx.x * 16;
  const int eMine = min(base + lr, E - 1);
  const int nrow = ei[eMine];       // source node j  (edge_index[0])
  const int ncol = ei[E + eMine];   // dest node i    (edge_index[1])
  if (hi == 0) sCol[lr] = ncol;

  // coord diff / dist for edge (base+lr)
  const float cdx = x[ncol * 3 + 0] - x[nrow * 3 + 0];
  const float cdy = x[ncol * 3 + 1] - x[nrow * 3 + 1];
  const float cdz = x[ncol * 3 + 2] - x[nrow * 3 + 2];
  const float dist = sqrtf(cdx * cdx + cdy * cdy + cdz * cdz);

  // ---- layer 1: m1 = silu([h_i | h_j | dist] @ We1 + be1), K padded 129->160
  v8f acc[4];
  #pragma unroll
  for (int nt = 0; nt < 4; ++nt) {
    float b = be1[nt * 16 + lr];
    #pragma unroll
    for (int v = 0; v < 8; ++v) acc[nt][v] = b;
  }
  #pragma unroll
  for (int kb = 0; kb < 4; ++kb) {
    const float* src = h + (long)((kb < 2) ? ncol : nrow) * 64;
    v16bf a = frag_from_f32(src, (kb & 1) * 32 + ko0);
    #pragma unroll
    for (int nt = 0; nt < 4; ++nt) {
      v16bf b = frag_from_bf16(We1T + (nt * 16 + lr) * 160, kb * 32 + ko0);
      acc[nt] = wmma_bf16(a, b, acc[nt]);
    }
  }
  { // K-block 4: only K=128 (dist) is non-zero
    v16bf a;
    #pragma unroll
    for (int i = 0; i < 16; ++i) a[i] = (__bf16)0.0f;
    if (hi == 0) a[0] = (__bf16)dist;
    #pragma unroll
    for (int nt = 0; nt < 4; ++nt) {
      v16bf b = frag_from_bf16(We1T + (nt * 16 + lr) * 160, 128 + ko0);
      acc[nt] = wmma_bf16(a, b, acc[nt]);
    }
  }
  // silu -> LDS (bf16, row-major 16x64) for layer-2 A fragments
  #pragma unroll
  for (int nt = 0; nt < 4; ++nt)
    #pragma unroll
    for (int v = 0; v < 8; ++v)
      sA[(v + 8 * hi) * 64 + nt * 16 + lr] = (__bf16)silu_f(acc[nt][v]);
  __syncthreads();

  // ---- layer 2: m2 = silu(m1 @ We2 + be2)
  v8f acc2[4];
  #pragma unroll
  for (int nt = 0; nt < 4; ++nt) {
    float b = be2[nt * 16 + lr];
    #pragma unroll
    for (int v = 0; v < 8; ++v) acc2[nt][v] = b;
  }
  #pragma unroll
  for (int kb = 0; kb < 2; ++kb) {
    v16bf a = frag_from_bf16(sA + lr * 64, kb * 32 + ko0);
    #pragma unroll
    for (int nt = 0; nt < 4; ++nt) {
      v16bf b = frag_from_bf16(We2T + (nt * 16 + lr) * 64, kb * 32 + ko0);
      acc2[nt] = wmma_bf16(a, b, acc2[nt]);
    }
  }
  #pragma unroll
  for (int nt = 0; nt < 4; ++nt)
    #pragma unroll
    for (int v = 0; v < 8; ++v) acc2[nt][v] = silu_f(acc2[nt][v]);

  // ---- gate = sigmoid(m2 @ Wg + bg); per-row reduction across 16 lanes
  float red[8];
  #pragma unroll
  for (int v = 0; v < 8; ++v) red[v] = 0.0f;
  #pragma unroll
  for (int nt = 0; nt < 4; ++nt) {
    float wg = Wg[nt * 16 + lr];
    #pragma unroll
    for (int v = 0; v < 8; ++v) red[v] += acc2[nt][v] * wg;
  }
  #pragma unroll
  for (int m = 1; m <= 8; m <<= 1)
    #pragma unroll
    for (int v = 0; v < 8; ++v) red[v] += __shfl_xor(red[v], m, 32);

  float gm[8];
  #pragma unroll
  for (int v = 0; v < 8; ++v) {
    int er = base + v + 8 * hi;
    float em = (er < E) ? emask[er] : 0.0f;
    gm[v] = sigmoid_f(red[v] + bg[0]) * em;
  }
  #pragma unroll
  for (int nt = 0; nt < 4; ++nt)
    #pragma unroll
    for (int v = 0; v < 8; ++v) acc2[nt][v] *= gm[v];   // gated m

  // ---- scatter-add m into m_aggr[col], and stash m (bf16) for coord branch
  __syncthreads();   // layer-2 reads of sA done
  #pragma unroll
  for (int nt = 0; nt < 4; ++nt)
    #pragma unroll
    for (int v = 0; v < 8; ++v)
      sA[(v + 8 * hi) * 64 + nt * 16 + lr] = (__bf16)acc2[nt][v];
  #pragma unroll
  for (int v = 0; v < 8; ++v) {
    int er = base + v + 8 * hi;
    if (er < E) {
      int dn = sCol[v + 8 * hi];
      #pragma unroll
      for (int nt = 0; nt < 4; ++nt)
        atomicAdd(&m_aggr[(long)dn * 64 + nt * 16 + lr], acc2[nt][v]);
    }
  }
  __syncthreads();

  // ---- coord weight: cw = silu(m @ Wc1 + bc1) @ Wc2 + bc2
  v8f acc3[4];
  #pragma unroll
  for (int nt = 0; nt < 4; ++nt) {
    float b = bc1[nt * 16 + lr];
    #pragma unroll
    for (int v = 0; v < 8; ++v) acc3[nt][v] = b;
  }
  #pragma unroll
  for (int kb = 0; kb < 2; ++kb) {
    v16bf a = frag_from_bf16(sA + lr * 64, kb * 32 + ko0);
    #pragma unroll
    for (int nt = 0; nt < 4; ++nt) {
      v16bf b = frag_from_bf16(Wc1T + (nt * 16 + lr) * 64, kb * 32 + ko0);
      acc3[nt] = wmma_bf16(a, b, acc3[nt]);
    }
  }
  float red2[8];
  #pragma unroll
  for (int v = 0; v < 8; ++v) red2[v] = 0.0f;
  #pragma unroll
  for (int nt = 0; nt < 4; ++nt) {
    float wc = Wc2[nt * 16 + lr];
    #pragma unroll
    for (int v = 0; v < 8; ++v) red2[v] += silu_f(acc3[nt][v]) * wc;
  }
  #pragma unroll
  for (int m = 1; m <= 8; m <<= 1)
    #pragma unroll
    for (int v = 0; v < 8; ++v) red2[v] += __shfl_xor(red2[v], m, 32);

  if (lr == 0) {
    #pragma unroll
    for (int v = 0; v < 8; ++v) sCw[v + 8 * hi] = red2[v] + bc2[0];
  }
  __syncthreads();

  if (hi == 0 && (base + lr) < E) {
    float cw = sCw[lr];
    atomicAdd(&x_out[(long)ncol * 3 + 0], cdx * cw);
    atomicAdd(&x_out[(long)ncol * 3 + 1], cdy * cw);
    atomicAdd(&x_out[(long)ncol * 3 + 2], cdz * cw);
  }
}

// ---------------------------------------------------------------------------
// Node kernel: one wave32 per 16-node tile
// ---------------------------------------------------------------------------
__global__ void __launch_bounds__(32)
egnn_node_kernel(const float* __restrict__ h, const float* __restrict__ m_aggr,
                 const __bf16* __restrict__ Wn1T, const float* __restrict__ bn1,
                 const __bf16* __restrict__ Wn2T, const float* __restrict__ bn2,
                 float* __restrict__ h_out, int N)
{
  __shared__ __bf16 sA[16 * 64];
  const int l  = threadIdx.x;
  const int hi = l >> 4;
  const int lr = l & 15;
  const int ko0 = hi ? 8 : 0;
  const int base = blockIdx.x * 16;
  const int node = min(base + lr, N - 1);

  v8f acc[4];
  #pragma unroll
  for (int nt = 0; nt < 4; ++nt) {
    float b = bn1[nt * 16 + lr];
    #pragma unroll
    for (int v = 0; v < 8; ++v) acc[nt][v] = b;
  }
  #pragma unroll
  for (int kb = 0; kb < 4; ++kb) {
    const float* src = ((kb < 2) ? h : m_aggr) + (long)node * 64;
    v16bf a = frag_from_f32(src, (kb & 1) * 32 + ko0);
    #pragma unroll
    for (int nt = 0; nt < 4; ++nt) {
      v16bf b = frag_from_bf16(Wn1T + (nt * 16 + lr) * 128, kb * 32 + ko0);
      acc[nt] = wmma_bf16(a, b, acc[nt]);
    }
  }
  #pragma unroll
  for (int nt = 0; nt < 4; ++nt)
    #pragma unroll
    for (int v = 0; v < 8; ++v)
      sA[(v + 8 * hi) * 64 + nt * 16 + lr] = (__bf16)silu_f(acc[nt][v]);
  __syncthreads();

  v8f acc2[4];
  #pragma unroll
  for (int nt = 0; nt < 4; ++nt) {
    float b = bn2[nt * 16 + lr];
    #pragma unroll
    for (int v = 0; v < 8; ++v) acc2[nt][v] = b;
  }
  #pragma unroll
  for (int kb = 0; kb < 2; ++kb) {
    v16bf a = frag_from_bf16(sA + lr * 64, kb * 32 + ko0);
    #pragma unroll
    for (int nt = 0; nt < 4; ++nt) {
      v16bf b = frag_from_bf16(Wn2T + (nt * 16 + lr) * 64, kb * 32 + ko0);
      acc2[nt] = wmma_bf16(a, b, acc2[nt]);
    }
  }
  // h_out = h + dh
  #pragma unroll
  for (int nt = 0; nt < 4; ++nt)
    #pragma unroll
    for (int v = 0; v < 8; ++v) {
      int r = base + v + 8 * hi;
      if (r < N) {
        long idx = (long)r * 64 + nt * 16 + lr;
        h_out[idx] = h[idx] + acc2[nt][v];
      }
    }
}

// ---------------------------------------------------------------------------
extern "C" void kernel_launch(void* const* d_in, const int* in_sizes, int n_in,
                              void* d_out, int out_size, void* d_ws, size_t ws_size,
                              hipStream_t stream) {
  const float* h     = (const float*)d_in[0];
  const float* x     = (const float*)d_in[1];
  const int*   ei    = (const int*)  d_in[2];
  const float* emask = (const float*)d_in[3];
  const float* We1 = (const float*)d_in[4];  const float* be1 = (const float*)d_in[5];
  const float* We2 = (const float*)d_in[6];  const float* be2 = (const float*)d_in[7];
  const float* Wg  = (const float*)d_in[8];  const float* bg  = (const float*)d_in[9];
  const float* Wn1 = (const float*)d_in[10]; const float* bn1 = (const float*)d_in[11];
  const float* Wn2 = (const float*)d_in[12]; const float* bn2 = (const float*)d_in[13];
  const float* Wc1 = (const float*)d_in[14]; const float* bc1 = (const float*)d_in[15];
  const float* Wc2 = (const float*)d_in[16]; const float* bc2 = (const float*)d_in[17];

  const int N = in_sizes[0] / 64;     // 50000
  const int E = in_sizes[2] / 2;      // 800000

  char* ws = (char*)d_ws;
  __bf16* We1T = (__bf16*)ws;             // 64 x 160
  __bf16* We2T = We1T + 64 * 160;         // 64 x 64
  __bf16* Wc1T = We2T + 64 * 64;          // 64 x 64
  __bf16* Wn1T = Wc1T + 64 * 64;          // 64 x 128
  __bf16* Wn2T = Wn1T + 64 * 128;         // 64 x 64
  float*  m_aggr = (float*)(ws + 65536);  // N x 64 fp32

  float* h_out = (float*)d_out;
  float* x_out = h_out + (long)N * 64;

  transpose_to_bf16<<<(64 * 160 + 255) / 256, 256, 0, stream>>>(We1, We1T, 129, 64, 160);
  transpose_to_bf16<<<(64 * 64  + 255) / 256, 256, 0, stream>>>(We2, We2T,  64, 64,  64);
  transpose_to_bf16<<<(64 * 64  + 255) / 256, 256, 0, stream>>>(Wc1, Wc1T,  64, 64,  64);
  transpose_to_bf16<<<(64 * 128 + 255) / 256, 256, 0, stream>>>(Wn1, Wn1T, 128, 64, 128);
  transpose_to_bf16<<<(64 * 64  + 255) / 256, 256, 0, stream>>>(Wn2, Wn2T,  64, 64,  64);

  long nm = (long)N * 64;
  init_buffers<<<(int)((nm + 255) / 256), 256, 0, stream>>>(m_aggr, nm, x_out, x, N * 3);

  int eTiles = (E + 15) / 16;
  egnn_edge_kernel<<<eTiles, 32, 0, stream>>>(h, x, ei, emask,
                                              We1T, be1, We2T, be2, Wg, bg,
                                              Wc1T, bc1, Wc2, bc2,
                                              m_aggr, x_out, E);

  int nTiles = (N + 15) / 16;
  egnn_node_kernel<<<nTiles, 32, 0, stream>>>(h, m_aggr, Wn1T, bn1, Wn2T, bn2, h_out, N);
}
